// GraphConv_67826123538776
// MI455X (gfx1250) — matrix operand order repaired and checked by reading, not compiled, not run on hardware
//
#include <hip/hip_runtime.h>
#include <hip/hip_bf16.h>

typedef __attribute__((ext_vector_type(16))) _Float16 v16h;
typedef __attribute__((ext_vector_type(8)))  _Float16 v8h;
typedef __attribute__((ext_vector_type(8)))  float    v8f;

#define N_NODES 8192
#define N_EDGES 262144
#define D_IN    512
#define D_OUT   512

// ---------------- workspace layout (bytes) ----------------
// adj  f16 : 8192*8192*2 = 134217728
// xwt  f16 : 512*8192*2  =   8388608   (XW^T, N-major: xwt[n*8192 + m])
// xh   f16 : 8192*512*2  =   8388608
// wT   f16 : 512*512*2   =    524288   (wT[n*512 + k] = W[k][n])
// dinv f32 : 8192*4      =     32768
#define OFF_ADJ  0
#define OFF_XWT  (134217728ull)
#define OFF_XH   (134217728ull + 8388608ull)
#define OFF_WT   (134217728ull + 8388608ull + 8388608ull)
#define OFF_DINV (134217728ull + 8388608ull + 8388608ull + 524288ull)

// ---------------- utility kernels ----------------
__global__ void zero_kernel(uint4* __restrict__ p, size_t n16) {
    size_t i = (size_t)blockIdx.x * blockDim.x + threadIdx.x;
    size_t stride = (size_t)gridDim.x * blockDim.x;
    uint4 z; z.x = z.y = z.z = z.w = 0u;
    for (; i < n16; i += stride) p[i] = z;
}

__global__ void scatter_kernel(const int* __restrict__ ei, _Float16* __restrict__ adj) {
    int e = blockIdx.x * blockDim.x + threadIdx.x;
    if (e >= N_EDGES) return;
    int i = ei[e];
    int j = ei[N_EDGES + e];
    adj[((size_t)i << 13) + j] = (_Float16)1.0f;   // idempotent "set" — races benign
    adj[((size_t)j << 13) + i] = (_Float16)1.0f;
}

__global__ void add_diag_kernel(_Float16* __restrict__ adj) {
    int i = blockIdx.x * blockDim.x + threadIdx.x;
    if (i >= N_NODES) return;
    size_t idx = ((size_t)i << 13) + i;
    adj[idx] = (_Float16)((float)adj[idx] + 1.0f);  // adj + eye (diag may become 2)
}

__global__ void degree_kernel(const _Float16* __restrict__ adj, float* __restrict__ dinv) {
    int row = blockIdx.x;
    const _Float16* r = adj + ((size_t)row << 13);
    float s = 0.0f;
    for (int t = threadIdx.x; t < N_NODES; t += 256) s += (float)r[t];
    __shared__ float red[256];
    red[threadIdx.x] = s;
    __syncthreads();
    for (int w = 128; w > 0; w >>= 1) {
        if ((int)threadIdx.x < w) red[threadIdx.x] += red[threadIdx.x + w];
        __syncthreads();
    }
    if (threadIdx.x == 0) dinv[row] = rsqrtf(red[0]);
}

__global__ void scale_kernel(_Float16* __restrict__ adj, const float* __restrict__ dinv) {
    size_t gid = (size_t)blockIdx.x * blockDim.x + threadIdx.x;   // one thread per 8 halves
    int row = (int)(gid >> 10);
    int c0  = (int)(gid & 1023) << 3;
    float di = dinv[row];
    _Float16* p = adj + ((size_t)row << 13) + c0;
    v8h v = *(const v8h*)p;
#pragma unroll
    for (int i = 0; i < 8; ++i)
        v[i] = (_Float16)((float)v[i] * di * dinv[c0 + i]);
    *(v8h*)p = v;
}

__global__ void cvt_x_kernel(const float* __restrict__ x, _Float16* __restrict__ xh) {
    size_t i = (size_t)blockIdx.x * blockDim.x + threadIdx.x;    // 8192*512 threads
    xh[i] = (_Float16)x[i];
}

__global__ void cvt_wT_kernel(const float* __restrict__ w, _Float16* __restrict__ wT) {
    int i = blockIdx.x * blockDim.x + threadIdx.x;               // 512*512 threads
    int n = i >> 9, k = i & 511;
    wT[(size_t)n * 512 + k] = (_Float16)w[(size_t)k * 512 + n];
}

// ---------------- WMMA fragment helpers ----------------
// A fragment (16x32 f16): lane M=lane%16; halves 0..7 = K[8*laneHi,+8),
// halves 8..15 = K[16+8*laneHi,+8)  -> two 16B loads.
__device__ __forceinline__ v16h load_afrag(const _Float16* __restrict__ arow, int k0) {
    v8h lo = *(const v8h*)(arow + k0);
    v8h hi = *(const v8h*)(arow + k0 + 16);
    v16h a;
#pragma unroll
    for (int i = 0; i < 8; ++i) { a[i] = lo[i]; a[i + 8] = hi[i]; }
    return a;
}

__device__ __forceinline__ v8f wmma_f16(v16h a, v16h b, v8f c) {
    return __builtin_amdgcn_wmma_f32_16x16x32_f16(false, a, false, b, (short)0, c,
                                                  false, false);
}

// 2x2 register-tiled GEMM body: one wave computes a 32x32 f32 block of
// A(MxK, f16 row-major) @ B(KxN) with B given N-major (BT[n*ldbt + k]).
// Per K-step: 8 b128 loads feed 4 WMMAs (2 loads/WMMA vs 4 untiled).
__device__ __forceinline__ void wmma_tile2x2(const _Float16* __restrict__ A, int lda,
                                             const _Float16* __restrict__ BT, int ldbt,
                                             int blkM, int blkN, int K, int lane,
                                             v8f& acc00, v8f& acc01,
                                             v8f& acc10, v8f& acc11) {
    const int laneLo = lane & 15;
    const int laneHi = lane >> 4;
    const _Float16* arow0 = A  + (size_t)(blkM * 32 + laneLo) * lda  + 8 * laneHi;
    const _Float16* arow1 = arow0 + (size_t)16 * lda;
    const _Float16* bcol0 = BT + (size_t)(blkN * 32 + laneLo) * ldbt + 8 * laneHi;
    const _Float16* bcol1 = bcol0 + (size_t)16 * ldbt;
    for (int k0 = 0; k0 < K; k0 += 32) {
        __builtin_prefetch(arow0 + k0 + 256, 0, 1);     // global_prefetch_b8 ahead on A
        __builtin_prefetch(arow1 + k0 + 256, 0, 1);
        v16h a0 = load_afrag(arow0, k0);
        v16h a1 = load_afrag(arow1, k0);
        v16h b0 = load_afrag(bcol0, k0);                // B: K=k0+8*laneHi+[0,8) and +16, N=laneLo
        v16h b1 = load_afrag(bcol1, k0);
        acc00 = wmma_f16(a0, b0, acc00);
        acc01 = wmma_f16(a0, b1, acc01);
        acc10 = wmma_f16(a1, b0, acc10);
        acc11 = wmma_f16(a1, b1, acc11);
    }
}

// GEMM-1:  XW^T = (x @ W)^T, f16 output stored N-major for GEMM-2's B loads.
__global__ void __launch_bounds__(256)
gemm_xw_kernel(const _Float16* __restrict__ xh, const _Float16* __restrict__ wT,
               _Float16* __restrict__ xwt) {
    int wave = (blockIdx.x * blockDim.x + threadIdx.x) >> 5;
    int lane = threadIdx.x & 31;
    int blkN = wave & 15;      // 512/32 = 16 N-blocks
    int blkM = wave >> 4;      // 8192/32 = 256 M-blocks
    v8f a00 = {}, a01 = {}, a10 = {}, a11 = {};
    wmma_tile2x2(xh, D_IN, wT, D_IN, blkM, blkN, D_IN, lane, a00, a01, a10, a11);
    // D layout: reg r <-> M = r + 8*laneHi, N = lane%16 (per 16x16 sub-tile)
    int laneLo = lane & 15;
    int moff   = (lane >> 4) << 3;
#pragma unroll
    for (int mb = 0; mb < 2; ++mb) {
#pragma unroll
        for (int nb = 0; nb < 2; ++nb) {
            const v8f& acc = mb == 0 ? (nb == 0 ? a00 : a01) : (nb == 0 ? a10 : a11);
            int n  = blkN * 32 + nb * 16 + laneLo;
            int m0 = blkM * 32 + mb * 16 + moff;
#pragma unroll
            for (int r = 0; r < 8; ++r)
                xwt[(size_t)n * N_NODES + m0 + r] = (_Float16)acc[r];
        }
    }
}

// GEMM-2:  out = adjS @ XW + bias   (dominant 68.7 GFLOP, f32 output)
__global__ void __launch_bounds__(256)
gemm_agg_kernel(const _Float16* __restrict__ adj, const _Float16* __restrict__ xwt,
                const float* __restrict__ bias, float* __restrict__ out) {
    int wave = (blockIdx.x * blockDim.x + threadIdx.x) >> 5;
    int lane = threadIdx.x & 31;
    int blkN = wave & 15;      // 16 N-blocks — block's waves share blkM -> A reuse
    int blkM = wave >> 4;      // 256 M-blocks
    v8f a00 = {}, a01 = {}, a10 = {}, a11 = {};
    wmma_tile2x2(adj, N_NODES, xwt, N_NODES, blkM, blkN, N_NODES, lane,
                 a00, a01, a10, a11);
    int laneLo = lane & 15;
    int moff   = (lane >> 4) << 3;
#pragma unroll
    for (int mb = 0; mb < 2; ++mb) {
#pragma unroll
        for (int nb = 0; nb < 2; ++nb) {
            const v8f& acc = mb == 0 ? (nb == 0 ? a00 : a01) : (nb == 0 ? a10 : a11);
            int n  = blkN * 32 + nb * 16 + laneLo;
            int m0 = blkM * 32 + mb * 16 + moff;
            float bv = bias[n];
#pragma unroll
            for (int r = 0; r < 8; ++r)
                out[(size_t)(m0 + r) * D_OUT + n] = acc[r] + bv;
        }
    }
}

// ---------------- launcher ----------------
extern "C" void kernel_launch(void* const* d_in, const int* in_sizes, int n_in,
                              void* d_out, int out_size, void* d_ws, size_t ws_size,
                              hipStream_t stream) {
    const float* x    = (const float*)d_in[0];
    const int*   ei   = (const int*)  d_in[1];
    const float* wgt  = (const float*)d_in[2];
    const float* bias = (const float*)d_in[3];
    float* out = (float*)d_out;

    char* ws = (char*)d_ws;
    _Float16* adj  = (_Float16*)(ws + OFF_ADJ);
    _Float16* xwt  = (_Float16*)(ws + OFF_XWT);
    _Float16* xh   = (_Float16*)(ws + OFF_XH);
    _Float16* wT   = (_Float16*)(ws + OFF_WT);
    float*    dinv = (float*)   (ws + OFF_DINV);

    // 1) adjacency build + normalize
    zero_kernel<<<4096, 256, 0, stream>>>((uint4*)adj, (size_t)N_NODES * N_NODES * 2 / 16);
    scatter_kernel<<<N_EDGES / 256, 256, 0, stream>>>(ei, adj);
    add_diag_kernel<<<N_NODES / 256, 256, 0, stream>>>(adj);
    degree_kernel<<<N_NODES, 256, 0, stream>>>(adj, dinv);
    scale_kernel<<<(N_NODES * (N_NODES / 8)) / 256, 256, 0, stream>>>(adj, dinv);

    // 2) feature transform: XW^T in f16 (N-major)
    cvt_x_kernel<<<(N_NODES * D_IN) / 256, 256, 0, stream>>>(x, xh);
    cvt_wT_kernel<<<(D_IN * D_OUT) / 256, 256, 0, stream>>>(wgt, wT);
    {
        int waves = (N_NODES / 32) * (D_OUT / 32);   // 4096
        gemm_xw_kernel<<<waves / 8, 256, 0, stream>>>(xh, wT, xwt);
    }

    // 3) aggregation GEMM + bias
    {
        int waves = (N_NODES / 32) * (D_OUT / 32);   // 4096
        gemm_agg_kernel<<<waves / 8, 256, 0, stream>>>(adj, xwt, bias, out);
    }
}
